// NonLocalBlock_13202729468161
// MI455X (gfx1250) — matrix operand order repaired and checked by reading, not compiled, not run on hardware
//
#include <hip/hip_runtime.h>
#include <hip/hip_bf16.h>
#include <math.h>

// Problem dims (fixed by the reference)
#define B_  4
#define C_  64
#define CI_ 32
#define N_  6400   // 80*80

#define LOG2E_ 1.4426950408889634f

typedef __bf16 bf16_t;
typedef __attribute__((ext_vector_type(8)))  __bf16 v8bf;
typedef __attribute__((ext_vector_type(16))) __bf16 v16bf;
typedef __attribute__((ext_vector_type(8)))  float  v8f;
typedef __attribute__((ext_vector_type(4)))  float  v4f;

__device__ __forceinline__ float exp2_fast(float x) {
  return __builtin_amdgcn_exp2f(x);   // v_exp_f32 (hardware exp2)
}

__device__ __forceinline__ v16bf cat8(v8bf lo, v8bf hi) {
  return __builtin_shufflevector(lo, hi, 0,1,2,3,4,5,6,7,8,9,10,11,12,13,14,15);
}

__device__ __forceinline__ v8f wmma_bf16(v16bf a, v16bf b, v8f c) {
  // D = A(16x32 bf16) * B(32x16 bf16) + C(16x16 f32)
  return __builtin_amdgcn_wmma_f32_16x16x32_bf16(false, a, false, b, (short)0, c,
                                                 false, false);
}

// ---------------------------------------------------------------------------
// Kernel 1: pointwise projections g,u,v = W*x + b.
//   u -> ubf  [B][N][CI] bf16, PRE-SCALED by log2(e) (scores live in exp2 domain)
//   v -> vbf  [B][N][CI] bf16  (m-major rows: WMMA B fragments, N=m, K=c)
//   g -> graw [B][CI][N] f32   (c-major: later scaled by 1/D[m] -> B frags, K=m)
// ---------------------------------------------------------------------------
__global__ void proj_kernel(const float* __restrict__ x,
                            const float* __restrict__ Wg, const float* __restrict__ bg,
                            const float* __restrict__ Wu, const float* __restrict__ bu,
                            const float* __restrict__ Wv, const float* __restrict__ bv,
                            float* __restrict__ graw,
                            bf16_t* __restrict__ ubf, bf16_t* __restrict__ vbf)
{
  int idx = blockIdx.x * blockDim.x + threadIdx.x;
  if (idx >= B_ * CI_ * N_) return;
  int n  = idx % N_;
  int t  = idx / N_;
  int ci = t % CI_;
  int b  = t / CI_;

  const float* xp = x + (size_t)b * C_ * N_ + n;
  const float* wg = Wg + ci * C_;
  const float* wu = Wu + ci * C_;
  const float* wv = Wv + ci * C_;
  float sg = bg[ci], su = bu[ci], sv = bv[ci];
#pragma unroll 8
  for (int c = 0; c < C_; ++c) {
    float xv = xp[(size_t)c * N_];
    sg = fmaf(wg[c], xv, sg);
    su = fmaf(wu[c], xv, su);
    sv = fmaf(wv[c], xv, sv);
  }
  graw[((size_t)b * CI_ + ci) * N_ + n] = sg;
  ubf[((size_t)b * N_ + n) * CI_ + ci]  = (bf16_t)(su * LOG2E_);  // exp2 domain
  vbf[((size_t)b * N_ + n) * CI_ + ci]  = (bf16_t)sv;
}

// ---------------------------------------------------------------------------
// Kernel 2 (pass 1): per-column softmax stats over n of S2[n,m] (exp2 domain).
// 4 waves per 16-column m-tile; wave w streams n-tiles w*16, w*16+64, ...
// Each n-tile is one v_wmma_f32_16x16x32_bf16 (K = CI = 32 exactly).
// D-layout: lane L holds column m = mbase+(L%16), rows r (+8 if L>=16).
// Per-wave stats combined across the 4 waves through LDS.
// ---------------------------------------------------------------------------
__global__ void colstats_kernel(const bf16_t* __restrict__ ubf,
                                const bf16_t* __restrict__ vbf,
                                float* __restrict__ Mstats, float* __restrict__ Dstats)
{
  __shared__ float Ms[4][16];
  __shared__ float Ds[4][16];

  const int blk   = blockIdx.x;
  const int b     = blk / (N_ / 16);
  const int mt    = blk % (N_ / 16);
  const int mbase = mt * 16;
  const int wave  = threadIdx.x >> 5;
  const int lane  = threadIdx.x & 31;
  const int row   = lane & 15;
  const int kb    = (lane < 16) ? 0 : 16;   // A-fragment K sub-chunk base
  const int koff  = (lane < 16) ? 0 : 16;   // B-fragment K chunk base

  // B fragment (fixed): v columns mbase..mbase+15, K=c
  const v16bf bfragv =
      *(const v16bf*)(vbf + ((size_t)b * N_ + mbase + row) * CI_ + koff);

  float Ml = -3.0e38f, Dl = 0.0f;
  const bf16_t* ub = ubf + (size_t)b * N_ * CI_;
  for (int nb = wave * 16; nb < N_; nb += 64) {
    const bf16_t* arow = ub + (size_t)(nb + row) * CI_;
    __builtin_prefetch(arow + 64 * CI_, 0, 1);   // global_prefetch next block
    v16bf afrag = cat8(*(const v8bf*)(arow + kb), *(const v8bf*)(arow + kb + 8));
    v8f z = {};
    v8f s = wmma_bf16(afrag, bfragv, z);

    float tmax = s[0];
#pragma unroll
    for (int r = 1; r < 8; ++r) tmax = fmaxf(tmax, s[r]);
    float newM = fmaxf(Ml, tmax);
    float acc = 0.0f;
#pragma unroll
    for (int r = 0; r < 8; ++r) acc += exp2_fast(s[r] - newM);
    Dl = Dl * exp2_fast(Ml - newM) + acc;
    Ml = newM;
  }
  // combine the two lane-halves holding the same column (rows 0-7 vs 8-15)
  float oM = __shfl_xor(Ml, 16, 32);
  float oD = __shfl_xor(Dl, 16, 32);
  float M  = fmaxf(Ml, oM);
  float D  = Dl * exp2_fast(Ml - M) + oD * exp2_fast(oM - M);
  if (lane < 16) { Ms[wave][row] = M; Ds[wave][row] = D; }
  __syncthreads();
  if (threadIdx.x < 16) {
    const int m = threadIdx.x;
    float Mt = fmaxf(fmaxf(Ms[0][m], Ms[1][m]), fmaxf(Ms[2][m], Ms[3][m]));
    float Dt = Ds[0][m] * exp2_fast(Ms[0][m] - Mt) + Ds[1][m] * exp2_fast(Ms[1][m] - Mt) +
               Ds[2][m] * exp2_fast(Ms[2][m] - Mt) + Ds[3][m] * exp2_fast(Ms[3][m] - Mt);
    Mstats[(size_t)b * N_ + mbase + m] = Mt;
    Dstats[(size_t)b * N_ + mbase + m] = Dt;
  }
}

// ---------------------------------------------------------------------------
// Kernel 3: g' = g / D  (bf16, c-major so m-chunks are contiguous B fragments)
// ---------------------------------------------------------------------------
__global__ void scale_g_kernel(const float* __restrict__ graw,
                               const float* __restrict__ Dstats,
                               bf16_t* __restrict__ gs)
{
  int idx = blockIdx.x * blockDim.x + threadIdx.x;
  if (idx >= B_ * CI_ * N_) return;
  int m = idx % N_;
  int b = idx / (CI_ * N_);
  gs[idx] = (bf16_t)(graw[idx] / Dstats[(size_t)b * N_ + m]);
}

// ---------------------------------------------------------------------------
// Kernel 4 (pass 2): fused  y[n,c] = sum_m exp2(S2[n,m]-M[m]) * g'[c,m]
// then out = Ww*y + bw + x.  Two waves per 16-row n-tile, K(m)-split:
// wave w handles m-chunks w*32, w*32+64, ...  Each chunk: 2 score WMMAs ->
// exp2 -> private LDS tile (D->A transpose) -> 2 accumulate WMMAs (K=m).
// Partial y accumulators summed through LDS; wave 0 runs the projection
// epilogue (4 more WMMAs) + bias + residual store.
// ---------------------------------------------------------------------------
__global__ void attn_out_kernel(const bf16_t* __restrict__ ubf,
                                const bf16_t* __restrict__ vbf,
                                const bf16_t* __restrict__ gs,
                                const float* __restrict__ Mstats,
                                const float* __restrict__ Ww,
                                const float* __restrict__ bw,
                                const float* __restrict__ x,
                                float* __restrict__ out)
{
  __shared__ __align__(64) bf16_t Pt[2][16 * 32];  // per-wave [n(16)][m or c(32)]
  __shared__ float Ycomb[32][16];                  // wave-1 partial y handoff

  const int blk   = blockIdx.x;
  const int b     = blk / (N_ / 16);
  const int nt    = blk % (N_ / 16);
  const int nbase = nt * 16;
  const int wave  = threadIdx.x >> 5;
  const int lane  = threadIdx.x & 31;
  const int row   = lane & 15;
  const int kb    = (lane < 16) ? 0 : 16;
  const int koff  = (lane < 16) ? 0 : 16;
  const int roff  = (lane < 16) ? 0 : 8;   // D-layout row offset

  // A fragment for score WMMAs (fixed: u rows nbase..nbase+15)
  const bf16_t* urow = ubf + ((size_t)b * N_ + nbase + row) * CI_;
  const v16bf au = cat8(*(const v8bf*)(urow + kb), *(const v8bf*)(urow + kb + 8));

  v8f acc0 = {};   // y columns c = 0..15
  v8f acc1 = {};   // y columns c = 16..31

  bf16_t* myPt = &Pt[wave][0];
  for (int mc = wave * 32; mc < N_; mc += 64) {
#pragma unroll
    for (int t = 0; t < 2; ++t) {
      const int mcol = mc + t * 16 + row;
      const v16bf bv =
          *(const v16bf*)(vbf + ((size_t)b * N_ + mcol) * CI_ + koff);
      v8f z = {};
      v8f s = wmma_bf16(au, bv, z);
      const float Mc = Mstats[(size_t)b * N_ + mcol];
#pragma unroll
      for (int r = 0; r < 8; ++r)
        myPt[(r + roff) * 32 + t * 16 + row] = (bf16_t)exp2_fast(s[r] - Mc);
    }
    __syncthreads();
    // A fragment of P tile (M=n, K=m)
    const v16bf pa = cat8(*(const v8bf*)(&myPt[row * 32 + kb]),
                          *(const v8bf*)(&myPt[row * 32 + kb + 8]));
    // B fragments of g' (K=m chunk, N=c)
    const v16bf bg0 =
        *(const v16bf*)(gs + ((size_t)b * CI_ + row) * N_ + mc + koff);
    const v16bf bg1 =
        *(const v16bf*)(gs + ((size_t)b * CI_ + 16 + row) * N_ + mc + koff);
    acc0 = wmma_bf16(pa, bg0, acc0);
    acc1 = wmma_bf16(pa, bg1, acc1);
    __syncthreads();
  }

  // Combine the two waves' partial y tiles through LDS
  if (wave == 1) {
#pragma unroll
    for (int r = 0; r < 8; ++r) {
      Ycomb[lane][r]     = acc0[r];
      Ycomb[lane][r + 8] = acc1[r];
    }
  }
  __syncthreads();
  if (wave == 0) {
#pragma unroll
    for (int r = 0; r < 8; ++r) {
      acc0[r] += Ycomb[lane][r];
      acc1[r] += Ycomb[lane][r + 8];
    }

    // Transpose y tile (16n x 32c) through LDS for the projection A fragment
#pragma unroll
    for (int r = 0; r < 8; ++r) {
      myPt[(r + roff) * 32 + row]      = (bf16_t)acc0[r];
      myPt[(r + roff) * 32 + 16 + row] = (bf16_t)acc1[r];
    }
    const v16bf ya = cat8(*(const v8bf*)(&myPt[row * 32 + kb]),
                          *(const v8bf*)(&myPt[row * 32 + kb + 8]));

    // Final projection: out[n,o] = sum_c y[n,c]*Ww[o,c] + bw[o] + x
#pragma unroll
    for (int t = 0; t < 4; ++t) {
      const int o = t * 16 + row;
      v16bf wfrag;
      const float* wr = Ww + (size_t)o * CI_ + koff;
#pragma unroll
      for (int k = 0; k < 16; ++k) wfrag[k] = (bf16_t)wr[k];
      v8f z = {};
      v8f od = wmma_bf16(ya, wfrag, z);

      const float  bias = bw[o];
      const size_t base = ((size_t)b * C_ + o) * N_ + nbase + roff;
      const v4f xv0 = *(const v4f*)(x + base);
      const v4f xv1 = *(const v4f*)(x + base + 4);
      v4f y0, y1;
#pragma unroll
      for (int r = 0; r < 4; ++r) {
        y0[r] = od[r]     + bias + xv0[r];
        y1[r] = od[r + 4] + bias + xv1[r];
      }
      *(v4f*)(out + base)     = y0;
      *(v4f*)(out + base + 4) = y1;
    }
  }
}

// ---------------------------------------------------------------------------
extern "C" void kernel_launch(void* const* d_in, const int* in_sizes, int n_in,
                              void* d_out, int out_size, void* d_ws, size_t ws_size,
                              hipStream_t stream) {
  (void)in_sizes; (void)n_in; (void)out_size; (void)ws_size;
  const float* x  = (const float*)d_in[0];
  const float* Wg = (const float*)d_in[1];
  const float* bg = (const float*)d_in[2];
  const float* Wu = (const float*)d_in[3];
  const float* bu = (const float*)d_in[4];
  const float* Wv = (const float*)d_in[5];
  const float* bv = (const float*)d_in[6];
  const float* Ww = (const float*)d_in[7];
  const float* bw = (const float*)d_in[8];
  float* out = (float*)d_out;

  // Workspace carve-out (~8.4 MB total)
  char* ws = (char*)d_ws;
  size_t off = 0;
  auto take = [&](size_t bytes) -> void* {
    void* p = ws + off;
    off += (bytes + 255) & ~(size_t)255;
    return p;
  };
  bf16_t* ubf  = (bf16_t*)take((size_t)B_ * N_ * CI_ * sizeof(bf16_t));
  bf16_t* vbf  = (bf16_t*)take((size_t)B_ * N_ * CI_ * sizeof(bf16_t));
  float*  graw = (float*) take((size_t)B_ * CI_ * N_ * sizeof(float));
  bf16_t* gs   = (bf16_t*)take((size_t)B_ * CI_ * N_ * sizeof(bf16_t));
  float*  Mst  = (float*) take((size_t)B_ * N_ * sizeof(float));
  float*  Dst  = (float*) take((size_t)B_ * N_ * sizeof(float));

  const int total = B_ * CI_ * N_;
  proj_kernel<<<(total + 255) / 256, 256, 0, stream>>>(x, Wg, bg, Wu, bu, Wv, bv,
                                                       graw, ubf, vbf);
  colstats_kernel<<<B_ * (N_ / 16), 128, 0, stream>>>(ubf, vbf, Mst, Dst);
  scale_g_kernel<<<(total + 255) / 256, 256, 0, stream>>>(graw, Dst, gs);
  attn_out_kernel<<<B_ * (N_ / 16), 64, 0, stream>>>(ubf, vbf, gs, Mst, Ww, bw,
                                                     x, out);
}
